// GraphTransformerEncoder_53498112639113
// MI455X (gfx1250) — matrix operand order repaired and checked by reading, not compiled, not run on hardware
//
#include <hip/hip_runtime.h>

typedef __attribute__((ext_vector_type(2)))  float    v2f;
typedef __attribute__((ext_vector_type(8)))  float    v8f;
typedef __attribute__((ext_vector_type(16))) _Float16 v16h;
typedef __attribute__((ext_vector_type(4)))  unsigned u32x4;
typedef __attribute__((ext_vector_type(4)))  int      i32x4;
typedef __attribute__((ext_vector_type(8)))  int      i32x8;

#ifndef USE_F32_WMMA
#if __has_builtin(__builtin_amdgcn_wmma_f32_16x16x4_f32)
#define USE_F32_WMMA 1
#else
#define USE_F32_WMMA 0
#endif
#endif

#ifndef USE_TDM
#if __has_builtin(__builtin_amdgcn_tensor_load_to_lds)
#define USE_TDM 1
#if __has_include(<hip/amd_detail/amd_gfx1250_TDM.h>)
#define TDM_ARGS6 1            // therock-10.0 headers -> 6-arg builtin
#else
#define TDM_ARGS6 0            // ROCm 7.2 -> 5-arg builtin
#endif
#else
#define USE_TDM 0
#endif
#endif

#define HC 128
#define NHEAD 4
#define CCH 32
#define RSQRT_C 0.17677669529663687f
#define LN_EPS 1e-5f

// ---------------------------------------------------------------------------
// monotonic float <-> uint mapping so segment-max can use hardware atomicMax
// ---------------------------------------------------------------------------
__device__ __forceinline__ unsigned floatFlip(float f) {
  unsigned b = __float_as_uint(f);
  unsigned mask = (unsigned)(-(int)(b >> 31)) | 0x80000000u;
  return b ^ mask;
}
__device__ __forceinline__ float floatUnflip(unsigned u) {
  unsigned mask = (u >> 31) ? 0x80000000u : 0xFFFFFFFFu;
  return __uint_as_float(u ^ mask);
}

__device__ __forceinline__ float waveReduceSum(float s) {
#pragma unroll
  for (int off = 16; off > 0; off >>= 1) s += __shfl_xor(s, off, 32);
  return s;
}

// ---------------------------------------------------------------------------
// Y[nrows,128] = X[nrows,128] @ W[128,128] + bias
// Wave = 16-row x 128-col strip, K=128.  W staged once per block into LDS
// (via the Tensor Data Mover when available), B-fragments fed from LDS.
// ---------------------------------------------------------------------------
__global__ __launch_bounds__(256) void gemm_bias_128(
    const float* __restrict__ X, const float* __restrict__ W,
    const float* __restrict__ bias, float* __restrict__ Y, int nrows) {
  __shared__ float wsm[HC * HC];                   // 64 KB of the 320 KB WGP LDS

  // ---- stage W[128,128] into LDS ----
#if USE_TDM
  if (threadIdx.x == 0) {
    unsigned lds_addr = (unsigned)(__attribute__((address_space(3))) float*)wsm;
    unsigned long long ga = (unsigned long long)W;
    // D# group0: count=1 | lds_addr | global_addr[56:0] | type=2
    u32x4 g0 = {1u, lds_addr, (unsigned)ga,
                (unsigned)((ga >> 32) & 0x01FFFFFFu) | (2u << 30)};
    // D# group1: data_size=4B; tensor_dim0=dim1=128; tile_dim0=tile_dim1=128;
    //            tensor_dim0_stride=128  (ISA 8.4 bit layout)
    i32x8 g1 = {(int)(2u << 16),      // workgroup_mask=0, data_size=2 (4B)
                (int)(128u << 16),    // [63:48] tensor_dim0 lo16
                (int)(128u << 16),    // [95:80] tensor_dim1 lo16
                (int)(128u << 16),    // [127:112] tile_dim0
                128,                  // [143:128] tile_dim1 (tile_dim2=0)
                128,                  // [191:160] tensor_dim0_stride lo32
                0, 0};
    i32x4 gz = {0, 0, 0, 0};          // groups 2/3: dims beyond 2D unused
#if TDM_ARGS6
    i32x8 z8 = {0, 0, 0, 0, 0, 0, 0, 0};
    __builtin_amdgcn_tensor_load_to_lds(g0, g1, gz, gz, z8, 0);
#else
    __builtin_amdgcn_tensor_load_to_lds(g0, g1, gz, gz, 0);
#endif
    __builtin_amdgcn_s_wait_tensorcnt(0);
  }
  __syncthreads();
#else
  {
    const float4* __restrict__ wv = (const float4*)W;
    float4* sv = (float4*)wsm;
    for (int i = threadIdx.x; i < (HC * HC) / 4; i += 256) sv[i] = wv[i];
    __syncthreads();
  }
#endif

  const int lane = threadIdx.x & 31;
  const int wave = threadIdx.x >> 5;
  const int m    = lane & 15;   // M index (A) / N index (B, C, D)
  const int half = lane >> 4;
  const int row0 = blockIdx.x * 128 + wave * 16;

  int arow = row0 + m;
  if (arow >= nrows) arow = nrows - 1;             // clamp; stores predicated
  const float* __restrict__ xrow = X + (size_t)arow * HC;

  v8f acc[8];
#pragma unroll
  for (int t = 0; t < 8; ++t) {
    float bv = bias[t * 16 + m];                   // bias depends on col only
#pragma unroll
    for (int r = 0; r < 8; ++r) acc[t][r] = bv;
  }

#if USE_F32_WMMA
  for (int k0 = 0; k0 < HC; k0 += 4) {
    // A 16x4 f32: lanes 0-15 hold K={k0,k0+1}, lanes 16-31 hold K={k0+2,k0+3}
    v2f a;
    a.x = xrow[k0 + 2 * half];
    a.y = xrow[k0 + 2 * half + 1];
    const float* wrow = &wsm[(k0 + 2 * half) * HC];
#pragma unroll
    for (int t = 0; t < 8; ++t) {
      v2f b;                                       // B 4x16: row K, col = t*16+m
      b.x = wrow[t * 16 + m];
      b.y = wrow[HC + t * 16 + m];
      acc[t] = __builtin_amdgcn_wmma_f32_16x16x4_f32(
          false, a, false, b, (short)0, acc[t], false, false);
    }
  }
#else
  for (int k0 = 0; k0 < HC; k0 += 32) {
    v16h a;                                        // A 16x32 f16 lane layout
#pragma unroll
    for (int j = 0; j < 8; ++j) {
      int K0 = k0 + (j >> 2) * 16 + half * 8 + (j & 3) * 2;
      a[2 * j]     = (_Float16)xrow[K0];
      a[2 * j + 1] = (_Float16)xrow[K0 + 1];
    }
#pragma unroll
    for (int t = 0; t < 8; ++t) {
      v16h b;                                      // B 32x16: same K order as A
#pragma unroll
      for (int j = 0; j < 8; ++j) {
        int K = k0 + (j >> 2) * 16 + half * 8 + (j & 3) * 2;
        b[2 * j]     = (_Float16)wsm[K * HC + t * 16 + m];
        b[2 * j + 1] = (_Float16)wsm[(K + 1) * HC + t * 16 + m];
      }
      acc[t] = __builtin_amdgcn_wmma_f32_16x16x32_f16(
          false, a, false, b, (short)0, acc[t], false, false);
    }
  }
#endif

#pragma unroll
  for (int t = 0; t < 8; ++t) {
#pragma unroll
    for (int r = 0; r < 8; ++r) {
      int orow = row0 + half * 8 + r;              // C/D: M = r + 8*half
      if (orow < nrows) Y[(size_t)orow * HC + t * 16 + m] = acc[t][r];
    }
  }
}

// ---------------------------------------------------------------------------
// per-edge attention logits + segment max  (wave per edge, lane = channel)
// ---------------------------------------------------------------------------
__global__ __launch_bounds__(256) void edge_alpha_kernel(
    const long long* __restrict__ ei, const float* __restrict__ q,
    const float* __restrict__ k, float* __restrict__ ex,
    unsigned* __restrict__ amax, int nedges) {
  const int lane = threadIdx.x & 31;
  const long long e = (long long)blockIdx.x * 8 + (threadIdx.x >> 5);
  if (e >= nedges) return;
  const long long s = ei[e], d = ei[nedges + e];
  const float* __restrict__ qd = q + (size_t)d * HC;
  const float* __restrict__ ks = k + (size_t)s * HC;
#pragma unroll
  for (int h = 0; h < NHEAD; ++h) {
    float p = qd[h * CCH + lane] * ks[h * CCH + lane];
    p = waveReduceSum(p);
    if (lane == 0) {
      float alpha = p * RSQRT_C;
      ex[e * NHEAD + h] = alpha;
      atomicMax(&amax[(size_t)d * NHEAD + h], floatFlip(alpha));
    }
  }
}

// ---------------------------------------------------------------------------
// exp(alpha - segmax) + segment sum   (thread per edge*head)
// ---------------------------------------------------------------------------
__global__ __launch_bounds__(256) void edge_exp_kernel(
    const long long* __restrict__ ei, float* __restrict__ ex,
    const unsigned* __restrict__ amax, float* __restrict__ denom, int nedges) {
  const long long i = (long long)blockIdx.x * blockDim.x + threadIdx.x;
  if (i >= (long long)nedges * NHEAD) return;
  const long long e = i >> 2;
  const int h = (int)(i & 3);
  const long long d = ei[nedges + e];
  float am = floatUnflip(amax[(size_t)d * NHEAD + h]);
  float val = expf(ex[i] - am);
  ex[i] = val;
  atomicAdd(&denom[(size_t)d * NHEAD + h], val);
}

// ---------------------------------------------------------------------------
// out[dst] += (ex/denom) * v[src]   (wave per edge, lane = channel)
// ---------------------------------------------------------------------------
__global__ __launch_bounds__(256) void edge_scatter_kernel(
    const long long* __restrict__ ei, const float* __restrict__ ex,
    const float* __restrict__ denom, const float* __restrict__ v,
    float* __restrict__ out, int nedges) {
  const int lane = threadIdx.x & 31;
  const long long e = (long long)blockIdx.x * 8 + (threadIdx.x >> 5);
  if (e >= nedges) return;
  const long long s = ei[e], d = ei[nedges + e];
  const float* __restrict__ vs = v + (size_t)s * HC;
  float* __restrict__ od = out + (size_t)d * HC;
#pragma unroll
  for (int h = 0; h < NHEAD; ++h) {
    float w = ex[e * NHEAD + h] / denom[(size_t)d * NHEAD + h];
    atomicAdd(&od[h * CCH + lane], w * vs[h * CCH + lane]);
  }
}

// ---------------------------------------------------------------------------
// layer norm (+ optional relu)   (wave per node, 4 channels per lane)
// ---------------------------------------------------------------------------
__global__ __launch_bounds__(256) void ln_kernel(
    const float* __restrict__ X, const float* __restrict__ g,
    const float* __restrict__ b, float* __restrict__ Y, int n, int do_relu) {
  const int lane = threadIdx.x & 31;
  const long long row = (long long)blockIdx.x * 8 + (threadIdx.x >> 5);
  if (row >= n) return;
  const float* __restrict__ xr = X + (size_t)row * HC;
  float val[4];
#pragma unroll
  for (int i = 0; i < 4; ++i) val[i] = xr[i * 32 + lane];
  float s = val[0] + val[1] + val[2] + val[3];
  s = waveReduceSum(s);
  float mu = s * (1.0f / HC);
  float q = 0.0f;
#pragma unroll
  for (int i = 0; i < 4; ++i) {
    val[i] -= mu;
    q += val[i] * val[i];
  }
  q = waveReduceSum(q);
  float inv = rsqrtf(q * (1.0f / HC) + LN_EPS);
  float* __restrict__ yr = Y + (size_t)row * HC;
#pragma unroll
  for (int i = 0; i < 4; ++i) {
    int c = i * 32 + lane;
    float y = val[i] * inv * g[c] + b[c];
    if (do_relu) y = fmaxf(y, 0.0f);
    yr[c] = y;
  }
}

// ---------------------------------------------------------------------------
extern "C" void kernel_launch(void* const* d_in, const int* in_sizes, int n_in,
                              void* d_out, int out_size, void* d_ws, size_t ws_size,
                              hipStream_t stream) {
  const int N = 50000, E = 800000;
  (void)in_sizes; (void)n_in; (void)out_size; (void)ws_size;

  const float* x = (const float*)d_in[0];
  const long long* ei = (const long long*)d_in[1];   // int64 [2,E]: row0=src, row1=dst
  const float *Wq1 = (const float*)d_in[2],  *bq1 = (const float*)d_in[3];
  const float *Wk1 = (const float*)d_in[4],  *bk1 = (const float*)d_in[5];
  const float *Wv1 = (const float*)d_in[6],  *bv1 = (const float*)d_in[7];
  const float *Ws1 = (const float*)d_in[8],  *bs1 = (const float*)d_in[9];
  const float *g1  = (const float*)d_in[10], *be1 = (const float*)d_in[11];
  const float *Wq2 = (const float*)d_in[12], *bq2 = (const float*)d_in[13];
  const float *Wk2 = (const float*)d_in[14], *bk2 = (const float*)d_in[15];
  const float *Wv2 = (const float*)d_in[16], *bv2 = (const float*)d_in[17];
  const float *Ws2 = (const float*)d_in[18], *bs2 = (const float*)d_in[19];
  const float *g2  = (const float*)d_in[20], *be2 = (const float*)d_in[21];

  // workspace layout (all f32 unless noted): 5*N*HC + E*4 + N*4(u32) + N*4
  float* ws = (float*)d_ws;
  const size_t NHC = (size_t)N * HC;
  float* q  = ws;
  float* k  = q + NHC;
  float* v  = k + NHC;
  float* o  = v + NHC;              // skip output, then atomically accumulated
  float* h  = o + NHC;              // layer-1 result (layer-2 input)
  float* ex = h + NHC;              // [E,4] logits -> exp weights
  unsigned* amax = (unsigned*)(ex + (size_t)E * NHEAD);  // [N,4] flipped-max
  float* denom   = (float*)(amax + (size_t)N * NHEAD);   // [N,4]

  const dim3 blk(256);
  const int gemm_grid = (N + 127) / 128;
  const int edge_grid = (E + 7) / 8;          // wave per edge
  const int node_grid = (N + 7) / 8;          // wave per node
  const int eh_grid   = (int)(((long long)E * NHEAD + 255) / 256);

  // ---------------- layer 1 ----------------
  gemm_bias_128<<<gemm_grid, blk, 0, stream>>>(x, Wq1, bq1, q, N);
  gemm_bias_128<<<gemm_grid, blk, 0, stream>>>(x, Wk1, bk1, k, N);
  gemm_bias_128<<<gemm_grid, blk, 0, stream>>>(x, Wv1, bv1, v, N);
  gemm_bias_128<<<gemm_grid, blk, 0, stream>>>(x, Ws1, bs1, o, N);
  hipMemsetAsync(amax, 0, (size_t)N * NHEAD * sizeof(unsigned), stream);
  hipMemsetAsync(denom, 0, (size_t)N * NHEAD * sizeof(float), stream);
  edge_alpha_kernel<<<edge_grid, blk, 0, stream>>>(ei, q, k, ex, amax, E);
  edge_exp_kernel<<<eh_grid, blk, 0, stream>>>(ei, ex, amax, denom, E);
  edge_scatter_kernel<<<edge_grid, blk, 0, stream>>>(ei, ex, denom, v, o, E);
  ln_kernel<<<node_grid, blk, 0, stream>>>(o, g1, be1, h, N, 1);

  // ---------------- layer 2 ----------------
  gemm_bias_128<<<gemm_grid, blk, 0, stream>>>(h, Wq2, bq2, q, N);
  gemm_bias_128<<<gemm_grid, blk, 0, stream>>>(h, Wk2, bk2, k, N);
  gemm_bias_128<<<gemm_grid, blk, 0, stream>>>(h, Wv2, bv2, v, N);
  gemm_bias_128<<<gemm_grid, blk, 0, stream>>>(h, Ws2, bs2, o, N);
  hipMemsetAsync(amax, 0, (size_t)N * NHEAD * sizeof(unsigned), stream);
  hipMemsetAsync(denom, 0, (size_t)N * NHEAD * sizeof(float), stream);
  edge_alpha_kernel<<<edge_grid, blk, 0, stream>>>(ei, q, k, ex, amax, E);
  edge_exp_kernel<<<eh_grid, blk, 0, stream>>>(ei, ex, amax, denom, E);
  edge_scatter_kernel<<<edge_grid, blk, 0, stream>>>(ei, ex, denom, v, o, E);
  ln_kernel<<<node_grid, blk, 0, stream>>>(o, g2, be2, (float*)d_out, N, 0);
}
